// SimpleEqAttentionHead_58110907515320
// MI455X (gfx1250) — matrix operand order repaired and checked by reading, not compiled, not run on hardware
//
#include <hip/hip_runtime.h>

// ---------------- problem dims ----------------
#define NATOMS 256
#define PPAR   2
#define LDEG   9
#define FDIM   64
#define ROWF   (NATOMS*PPAR*LDEG*FDIM)   // 294912 floats per 'a' row of x / out
#define NSLICE (NATOMS*PPAR*LDEG)        // 4608 (b,p,l) slices of 64 features
#define PITCH  72                        // 64 + 8 bf16 pad (144B row = 36 banks)
#define XPITCH 264                       // 256 + 8 bf16 pad
#define SLICES_PER_WG 16

typedef __bf16 v16bf __attribute__((ext_vector_type(16)));
typedef __bf16 v8bf  __attribute__((ext_vector_type(8)));
typedef float  v8f   __attribute__((ext_vector_type(8)));

union Frag { v16bf v; v8bf h[2]; };

__device__ __forceinline__ v8f wmma_bf16(const Frag& a, const Frag& b, v8f c) {
  return __builtin_amdgcn_wmma_f32_16x16x32_bf16(false, a.v, false, b.v,
                                                 (short)0, c, false, false);
}

// A fragment (16x32 MxK, 16-bit): lane row = lane%16; lane<16 -> K {0..7,16..23},
// lane>=16 -> K {8..15,24..31}: two 16B chunks, 32B apart.
__device__ __forceinline__ Frag fragA(const __bf16* base) {
  Frag f;
  f.h[0] = *(const v8bf*)(base);
  f.h[1] = *(const v8bf*)(base + 16);
  return f;
}
// B fragment (32x16 KxN, 16-bit) read from B^T stored row-major:
// lane col = lane%16; lane<16 -> K {0..7,16..23}, lane>=16 -> K {8..15,24..31}.
__device__ __forceinline__ Frag fragB(const __bf16* base) {
  Frag f;
  f.h[0] = *(const v8bf*)(base);
  f.h[1] = *(const v8bf*)(base + 16);
  return f;
}

__device__ __forceinline__ void atomic_add_f32(float* p, float v) {
  unsafeAtomicAdd(p, v);   // global_atomic_add_f32
}

// ---------------- pass 0: tiny prep ----------------
// MT[p][n][f]  = M[p][f][n] = sum_g Wq[p][f][g]*Wk[p][n][g]   (B^T of M, bf16)
// WvT[p][n][g] = Wv[p][g][n]                                   (B^T of Wv, bf16)
// S zeroed for the atomic accumulation.
__global__ __launch_bounds__(256) void prep_kernel(
    const float* __restrict__ Wq, const float* __restrict__ Wk,
    const float* __restrict__ Wv, __bf16* __restrict__ MT,
    __bf16* __restrict__ WvT, float* __restrict__ S)
{
  const int tid = blockIdx.x * 256 + threadIdx.x;        // 8192 threads
  const int p = tid >> 12, n = (tid >> 6) & 63, f = tid & 63;
  const float* wq = Wq + p*4096 + f*64;
  const float* wk = Wk + p*4096 + n*64;
  float acc = 0.f;
  #pragma unroll 8
  for (int g = 0; g < 64; ++g) acc += wq[g] * wk[g];
  MT[tid]  = (__bf16)acc;
  WvT[tid] = (__bf16)Wv[p*4096 + f*64 + n];
  for (int i = tid; i < NATOMS*NATOMS; i += 8192) S[i] = 0.f;
}

// ---------------- pass 1: scores S[a,c] = sum_slice X M X^T ----------------
// grid = (288 K-chunks, 4 C-block combos), block = 256 (8 waves)
__global__ __launch_bounds__(256) void dotprod_kernel(
    const float* __restrict__ x, const __bf16* __restrict__ MT,
    float* __restrict__ S)
{
  __shared__ __align__(16) __bf16 sM [2*64*PITCH];
  __shared__ __align__(16) __bf16 sXr[128*PITCH];
  __shared__ __align__(16) __bf16 sXc[128*PITCH];
  __shared__ __align__(16) __bf16 sY [128*PITCH];

  const int tid = threadIdx.x;
  const int lane = tid & 31, wid = tid >> 5;
  const int l16 = lane & 15, lh = lane >> 4;
  const int R0 = (blockIdx.y >> 1) * 128;     // 'a' rows of this C block
  const int C0 = (blockIdx.y & 1) * 128;      // 'c' cols of this C block
  const int wr = wid & 3, wc = wid >> 2;      // wave tile: rows wr*32, cols wc*64

  for (int i = tid; i < 2*64*64; i += 256) {  // stage M^T (both parities)
    int pp = i >> 12, r = (i >> 6) & 63, c = i & 63;
    sM[pp*64*PITCH + r*PITCH + c] = MT[i];
  }

  v8f acc[8] = {};                            // 2x4 tiles of 16x16 per wave

  for (int ss = 0; ss < SLICES_PER_WG; ++ss) {
    const int s = blockIdx.x * SLICES_PER_WG + ss;
    const int p = (s / LDEG) & 1;
    __syncthreads();                          // LDS reuse fence
    #pragma unroll
    for (int i = 0; i < 8; ++i) {             // stage both 128x64 x-tiles as bf16
      int fid = tid + 256*i;                  // float4 id
      int row = fid >> 4, c4 = fid & 15;
      float4 vr = *(const float4*)(x + (size_t)(R0+row)*ROWF + s*64 + c4*4);
      __bf16* dr = &sXr[row*PITCH + c4*4];
      dr[0]=(__bf16)vr.x; dr[1]=(__bf16)vr.y; dr[2]=(__bf16)vr.z; dr[3]=(__bf16)vr.w;
      float4 vc = *(const float4*)(x + (size_t)(C0+row)*ROWF + s*64 + c4*4);
      __bf16* dc = &sXc[row*PITCH + c4*4];
      dc[0]=(__bf16)vc.x; dc[1]=(__bf16)vc.y; dc[2]=(__bf16)vc.z; dc[3]=(__bf16)vc.w;
    }
    __syncthreads();
    // GEMM1: Y[128x64] = Xr @ M[p]; wave owns row-tile 'wid'
    #pragma unroll
    for (int tc = 0; tc < 4; ++tc) {
      v8f y = {};
      #pragma unroll
      for (int kk = 0; kk < 2; ++kk) {
        Frag a = fragA(&sXr[(wid*16 + l16)*PITCH + kk*32 + lh*8]);
        Frag b = fragB(&sM[p*64*PITCH + (tc*16 + l16)*PITCH + kk*32 + lh*8]);
        y = wmma_bf16(a, b, y);
      }
      #pragma unroll
      for (int i = 0; i < 8; ++i)             // D layout: row = i + 8*(lane/16)
        sY[(wid*16 + i + 8*lh)*PITCH + tc*16 + l16] = (__bf16)y[i];
    }
    __syncthreads();
    // GEMM2: acc += Y @ Xc^T  (K = 64 features)
    #pragma unroll
    for (int kk = 0; kk < 2; ++kk) {
      Frag a0 = fragA(&sY[(wr*32 +  0 + l16)*PITCH + kk*32 + lh*8]);
      Frag a1 = fragA(&sY[(wr*32 + 16 + l16)*PITCH + kk*32 + lh*8]);
      #pragma unroll
      for (int tc = 0; tc < 4; ++tc) {
        Frag b = fragB(&sXc[(wc*64 + tc*16 + l16)*PITCH + kk*32 + lh*8]);
        acc[tc]     = wmma_bf16(a0, b, acc[tc]);
        acc[4 + tc] = wmma_bf16(a1, b, acc[4 + tc]);
      }
    }
  }
  #pragma unroll
  for (int t = 0; t < 8; ++t) {               // fold partial C into global S
    int trr = t >> 2, tc = t & 3;
    int r0 = R0 + wr*32 + trr*16 + 8*lh;
    int c  = C0 + wc*64 + tc*16 + l16;
    #pragma unroll
    for (int i = 0; i < 8; ++i)
      atomic_add_f32(&S[(size_t)(r0 + i)*NATOMS + c], acc[t][i]);
  }
}

// ---------------- pass 2: softmax over axis 'a' (axis=-2), emit bf16 W ----------
__global__ __launch_bounds__(256) void softmax_kernel(
    const float* __restrict__ S, __bf16* __restrict__ Wbf)
{
  __shared__ float red[256];
  const int c = blockIdx.x, a = threadIdx.x;
  float v = S[(size_t)a*NATOMS + c];
  red[a] = v; __syncthreads();
  for (int o = 128; o > 0; o >>= 1) {
    if (a < o) red[a] = fmaxf(red[a], red[a+o]);
    __syncthreads();
  }
  float m = red[0]; __syncthreads();
  float e = __expf(v - m);
  red[a] = e; __syncthreads();
  for (int o = 128; o > 0; o >>= 1) {
    if (a < o) red[a] += red[a+o];
    __syncthreads();
  }
  Wbf[(size_t)a*NATOMS + c] = (__bf16)(e / red[0]);
}

// ---------------- pass 3: out[:,j,:] = (W @ x[:,j,:]) @ Wv[p] ----------------
// grid = 4608 slices, block = 256 (8 waves); W A-frags streamed from L2-hot bf16.
__global__ __launch_bounds__(256) void out_kernel(
    const float* __restrict__ x, const __bf16* __restrict__ Wbf,
    const __bf16* __restrict__ WvT, float* __restrict__ out)
{
  __shared__ __align__(16) __bf16 sXT[64*XPITCH];   // X_j^T : [g][b]
  __shared__ __align__(16) __bf16 sT [256*PITCH];   // T = W @ X_j : [a][g]
  const int tid = threadIdx.x;
  const int lane = tid & 31, wid = tid >> 5;
  const int l16 = lane & 15, lh = lane >> 4;
  const int j = blockIdx.x;
  const int p = (j / LDEG) & 1;

  #pragma unroll
  for (int i = 0; i < 16; ++i) {              // load + transpose X_j (256x64)
    int fid = tid + 256*i;
    int bb = fid >> 4, c4 = fid & 15;
    float4 v = *(const float4*)(x + (size_t)bb*ROWF + (size_t)j*64 + c4*4);
    sXT[(c4*4+0)*XPITCH + bb] = (__bf16)v.x;
    sXT[(c4*4+1)*XPITCH + bb] = (__bf16)v.y;
    sXT[(c4*4+2)*XPITCH + bb] = (__bf16)v.z;
    sXT[(c4*4+3)*XPITCH + bb] = (__bf16)v.w;
  }
  __syncthreads();

  // T = W @ X_j : 256x64, K=256; wave owns rows [wid*32, wid*32+32)
  v8f accT[8] = {};
  #pragma unroll
  for (int kk = 0; kk < 8; ++kk) {
    Frag a0 = fragA(&Wbf[(size_t)(wid*32 +  0 + l16)*NATOMS + kk*32 + lh*8]);
    Frag a1 = fragA(&Wbf[(size_t)(wid*32 + 16 + l16)*NATOMS + kk*32 + lh*8]);
    #pragma unroll
    for (int tc = 0; tc < 4; ++tc) {
      Frag b = fragB(&sXT[(tc*16 + l16)*XPITCH + kk*32 + lh*8]);
      accT[tc]     = wmma_bf16(a0, b, accT[tc]);
      accT[4 + tc] = wmma_bf16(a1, b, accT[4 + tc]);
    }
  }
  #pragma unroll
  for (int t = 0; t < 8; ++t) {
    int trr = t >> 2, tc = t & 3;
    #pragma unroll
    for (int i = 0; i < 8; ++i)
      sT[(wid*32 + trr*16 + i + 8*lh)*PITCH + tc*16 + l16] = (__bf16)accT[t][i];
  }
  __syncthreads();

  // Out_j = T @ Wv[p] : 256x64, K=64; f32 straight to HBM
  #pragma unroll
  for (int trr = 0; trr < 2; ++trr) {
    v8f acc[4] = {};
    #pragma unroll
    for (int kk = 0; kk < 2; ++kk) {
      Frag a = fragA(&sT[(wid*32 + trr*16 + l16)*PITCH + kk*32 + lh*8]);
      #pragma unroll
      for (int tc = 0; tc < 4; ++tc) {
        Frag b = fragB(&WvT[p*4096 + (tc*16 + l16)*64 + kk*32 + lh*8]);
        acc[tc] = wmma_bf16(a, b, acc[tc]);
      }
    }
    #pragma unroll
    for (int tc = 0; tc < 4; ++tc) {
      int rbase = wid*32 + trr*16 + 8*lh;
      #pragma unroll
      for (int i = 0; i < 8; ++i)
        out[(size_t)(rbase + i)*ROWF + (size_t)j*64 + tc*16 + l16] = acc[tc][i];
    }
  }
}

// ---------------- host launcher ----------------
extern "C" void kernel_launch(void* const* d_in, const int* in_sizes, int n_in,
                              void* d_out, int out_size, void* d_ws, size_t ws_size,
                              hipStream_t stream) {
  const float* x  = (const float*)d_in[0];
  const float* Wq = (const float*)d_in[1];
  const float* Wk = (const float*)d_in[2];
  const float* Wv = (const float*)d_in[3];
  float* out = (float*)d_out;

  char* ws = (char*)d_ws;                    // < 0.5 MB total
  __bf16* MT  = (__bf16*)(ws);               // 16 KB  : M^T bf16, per parity
  __bf16* WvT = (__bf16*)(ws + 16384);       // 16 KB  : Wv^T bf16, per parity
  float*  S   = (float* )(ws + 32768);       // 256 KB : raw scores
  __bf16* Wbf = (__bf16*)(ws + 294912);      // 128 KB : softmaxed weights bf16

  prep_kernel   <<<32, 256, 0, stream>>>(Wq, Wk, Wv, MT, WvT, S);
  dotprod_kernel<<<dim3(NSLICE / SLICES_PER_WG, 4), 256, 0, stream>>>(x, MT, S);
  softmax_kernel<<<NATOMS, 256, 0, stream>>>(S, Wbf);
  out_kernel    <<<NSLICE, 256, 0, stream>>>(x, Wbf, WvT, out);
}
